// SlotPool_82935818486351
// MI455X (gfx1250) — matrix operand order repaired and checked by reading, not compiled, not run on hardware
//
#include <hip/hip_runtime.h>
#include <hip/hip_bf16.h>
#include <math.h>

typedef __attribute__((ext_vector_type(16))) _Float16 v16h;
typedef __attribute__((ext_vector_type(8)))  _Float16 h8;
typedef __attribute__((ext_vector_type(4)))  _Float16 h4;
typedef __attribute__((ext_vector_type(8)))  float    v8f;
typedef __attribute__((ext_vector_type(4)))  float    f32x4;

union AF { v16h v; f32x4 q[2]; };

// Problem constants
#define BB   4096
#define NN   196
#define DD   32
#define MM   12
#define NHH  4
#define RPBS 208   // padded rpb row stride (floats), rows n>=196 are zero
// Output offsets (floats)
#define OFF_Z   0L
#define OFF_AM  131072L
#define OFF_HE  9764864L
#define OFF_S   9961472L
#define OFF_TK  11534336L

__device__ __forceinline__ float wred_max(float v) {
#pragma unroll
  for (int o = 16; o; o >>= 1) v = fmaxf(v, __shfl_xor(v, o, 32));
  return v;
}
__device__ __forceinline__ float wred_sum(float v) {
#pragma unroll
  for (int o = 16; o; o >>= 1) v += __shfl_xor(v, o, 32);
  return v;
}

// ---------------------------------------------------------------------------
// Kernel 0: batch-independent precompute.
//   qk   [48][32] f16  : (1/sqrt(8)) * Qp(head h) . Wk   (folded QK operand)
//   wv16 [32][32] f16  : Wv row-major (B-operand layout [e][c])
//   rpbp [48][208] f32 : rpb gathered per (j=h*12+m, n); n>=196 zero-padded
// ---------------------------------------------------------------------------
__global__ void precompute_kernel(const float* __restrict__ Q,
                                  const float* __restrict__ Wq,
                                  const float* __restrict__ Wk,
                                  const float* __restrict__ Wv,
                                  const float* __restrict__ qc,
                                  const float* __restrict__ kc,
                                  const float* __restrict__ rpb_bias,
                                  _Float16* __restrict__ qk,
                                  _Float16* __restrict__ wv16,
                                  float* __restrict__ rpbp) {
  __shared__ float Qp[MM * DD];
  const int tid = threadIdx.x;
  for (int t = tid; t < MM * DD; t += blockDim.x) {
    int m = t >> 5, f = t & 31;
    float s = 0.f;
    for (int g = 0; g < DD; ++g) s += Q[m * DD + g] * Wq[f * DD + g];
    Qp[t] = s;
  }
  __syncthreads();
  const float rs = 0.35355339059327373f;  // 1/sqrt(8)
  for (int t = tid; t < 48 * DD; t += blockDim.x) {
    int j = t >> 5, c = t & 31;
    int h = j / MM, m = j % MM;
    float s = 0.f;
    for (int e = 0; e < 8; ++e) s += Qp[m * DD + h * 8 + e] * Wk[(h * 8 + e) * DD + c];
    qk[t] = (_Float16)(s * rs);
  }
  for (int t = tid; t < DD * DD; t += blockDim.x) wv16[t] = (_Float16)Wv[t];
  for (int t = tid; t < 48 * RPBS; t += blockDim.x) {
    int j = t / RPBS, n = t % RPBS;
    float val = 0.f;
    if (n < NN) {
      int h = j / MM, m = j % MM;
      float dy = qc[m * 2 + 0] - kc[n * 2 + 0];
      float dx = qc[m * 2 + 1] - kc[n * 2 + 1];
      int iy = (int)fminf(fmaxf(rintf(dy) + 13.f, 0.f), 26.f);
      int ix = (int)fminf(fmaxf(rintf(dx) + 13.f, 0.f), 26.f);
      val = rpb_bias[h * 729 + iy * 27 + ix];
    }
    rpbp[t] = val;
  }
}

// ---------------------------------------------------------------------------
// Kernel 1: one workgroup (128 threads, 4 wave32) per batch element.
// ---------------------------------------------------------------------------
__global__ void __launch_bounds__(128)
slotpool_main(const float* __restrict__ T,
              const float* __restrict__ Wp,
              const float* __restrict__ bp,
              const _Float16* __restrict__ qk,
              const _Float16* __restrict__ wv16,
              const float* __restrict__ rpbp,
              float* __restrict__ out) {
  __shared__ __align__(16) _Float16 T16[208 * 32];    // T_b, f16; rows 196..207 zero
  __shared__ __align__(16) _Float16 A16[48 * 224];    // logits -> attention, f16
  __shared__ __align__(16) _Float16 V16T[32 * 224];   // V transposed [e][n], f16
  __shared__ float Aavg[MM * NN];
  __shared__ float Slds[MM * DD];
  __shared__ float Zacc[DD];

  const int b    = blockIdx.x;
  const int tid  = threadIdx.x;
  const int wave = tid >> 5;
  const int lane = tid & 31;
  const int lh   = lane & 15;
  const int hi   = lane >> 4;

  // ---- Phase 0: stage T (vectorized), zero pads, init accumulators ------
  const float* Tb = T + (size_t)b * (NN * DD);
  for (int i4 = tid; i4 < 1568; i4 += 128) {          // 1568 float4 = 196*32
    f32x4 v = ((const f32x4*)Tb)[i4];
    h4 hv; hv[0] = (_Float16)v[0]; hv[1] = (_Float16)v[1];
    hv[2] = (_Float16)v[2]; hv[3] = (_Float16)v[3];
    ((h4*)T16)[i4] = hv;
  }
  for (int i = tid; i < 12 * 32; i += 128)            // rows 196..207 -> 0
    T16[196 * 32 + i] = (_Float16)0.f;
  for (int i = tid; i < 48 * 16; i += 128)            // A16  n in [208,224)
    A16[(i / 16) * 224 + 208 + (i & 15)] = (_Float16)0.f;
  for (int i = tid; i < 32 * 16; i += 128)            // V16T n in [208,224)
    V16T[(i / 16) * 224 + 208 + (i & 15)] = (_Float16)0.f;
  for (int i = tid; i < MM * NN; i += 128) Aavg[i] = 0.f;
  if (tid < DD) Zacc[tid] = bp[tid];
  __syncthreads();

  // ---- Phase 1: GEMM1  logits[n][j] = T16 . qk^T  (K=32, 1 WMMA/tile) ---
  // Unconditional packed epilogue: rows n>=196 compute to exactly 0
  // (zero T rows, zero rpb padding) == required K-padding for GEMM2.
  for (int t = wave; t < 39; t += 4) {
    int mt = t % 13, jt = t / 13;
    AF a, bf;
    const _Float16* ap = &T16[(mt * 16 + lh) * 32 + hi * 16];
    a.q[0] = *(const f32x4*)(ap);
    a.q[1] = *(const f32x4*)(ap + 8);
    const _Float16* bp0 = qk + (jt * 16 + lh) * 32 + hi * 16;
    bf.q[0] = *(const f32x4*)(bp0);
    bf.q[1] = *(const f32x4*)(bp0 + 8);
    int n0 = mt * 16 + hi * 8;
    int j  = jt * 16 + lh;
    const f32x4* rp = (const f32x4*)&rpbp[j * RPBS + n0];
    f32x4 r0 = rp[0], r1 = rp[1];
    v8f c = {0.f, 0.f, 0.f, 0.f, 0.f, 0.f, 0.f, 0.f};
    c = __builtin_amdgcn_wmma_f32_16x16x32_f16(false, a.v, false, bf.v,
                                               (short)0, c, false, false);
    h8 st;
    st[0] = (_Float16)(c[0] + r0[0]); st[1] = (_Float16)(c[1] + r0[1]);
    st[2] = (_Float16)(c[2] + r0[2]); st[3] = (_Float16)(c[3] + r0[3]);
    st[4] = (_Float16)(c[4] + r1[0]); st[5] = (_Float16)(c[5] + r1[1]);
    st[6] = (_Float16)(c[6] + r1[2]); st[7] = (_Float16)(c[7] + r1[3]);
    *(h8*)&A16[j * 224 + n0] = st;                    // one ds_store_b128
  }
  // ---- Phase 1b: V = T . Wv^T -> stored transposed V16T[e][n] -----------
  for (int t = wave; t < 26; t += 4) {
    int mt = t % 13, et = t / 13;
    AF a, bf;
    const _Float16* ap = &T16[(mt * 16 + lh) * 32 + hi * 16];
    a.q[0] = *(const f32x4*)(ap);
    a.q[1] = *(const f32x4*)(ap + 8);
    const _Float16* bp0 = wv16 + (et * 16 + lh) * 32 + hi * 16;
    bf.q[0] = *(const f32x4*)(bp0);
    bf.q[1] = *(const f32x4*)(bp0 + 8);
    v8f c = {0.f, 0.f, 0.f, 0.f, 0.f, 0.f, 0.f, 0.f};
    c = __builtin_amdgcn_wmma_f32_16x16x32_f16(false, a.v, false, bf.v,
                                               (short)0, c, false, false);
    int n0 = mt * 16 + hi * 8;
    int e  = et * 16 + lh;
    h8 st;
#pragma unroll
    for (int v = 0; v < 8; ++v) st[v] = (_Float16)c[v];
    *(h8*)&V16T[e * 224 + n0] = st;                   // one ds_store_b128
  }
  __syncthreads();

  // ---- Phase 2: softmax over n per column j (wave w owns head h=w) ------
  const float invlogN = 1.0f / 5.2781147f;  // 1/ln(196)
  for (int jj = 0; jj < MM; ++jj) {
    int j = wave * MM + jj;
    float x[7];
#pragma unroll
    for (int i = 0; i < 7; ++i) {
      int n = lane + 32 * i;
      x[i] = (n < NN) ? (float)A16[j * 224 + n] : -1e30f;
    }
    float mx = x[0];
#pragma unroll
    for (int i = 1; i < 7; ++i) mx = fmaxf(mx, x[i]);
    mx = wred_max(mx);
    float s = 0.f;
#pragma unroll
    for (int i = 0; i < 7; ++i) {
      int n = lane + 32 * i;
      if (n < NN) { x[i] = __expf(x[i] - mx); s += x[i]; }
    }
    s = wred_sum(s);
    float inv = 1.0f / s;
    float ent = 0.f;
#pragma unroll
    for (int i = 0; i < 7; ++i) {
      int n = lane + 32 * i;
      if (n < NN) {
        float a = x[i] * inv;
        A16[j * 224 + n] = (_Float16)a;
        float p = fmaxf(a, 1e-12f);
        ent += p * __logf(p);
        atomicAdd(&Aavg[jj * NN + n], 0.25f * a);
      }
    }
    ent = wred_sum(ent);
    if (lane == 0)
      out[OFF_HE + (long)b * 48 + jj * 4 + wave] = 1.0f + ent * invlogN;
  }
  __syncthreads();

  // ---- Phase 3: GEMM2  S[j][e] = A^T . V  (K = 224, zero-padded) --------
  for (int t = wave; t < 6; t += 4) {
    int jt = t >> 1, et = t & 1;
    v8f acc = {0.f, 0.f, 0.f, 0.f, 0.f, 0.f, 0.f, 0.f};
#pragma unroll
    for (int kt = 0; kt < 7; ++kt) {
      int n0 = kt * 32;
      AF a, bf;
      const _Float16* ap = &A16[(jt * 16 + lh) * 224 + n0 + hi * 16];
      a.q[0] = *(const f32x4*)(ap);
      a.q[1] = *(const f32x4*)(ap + 8);
      const _Float16* vp = &V16T[(et * 16 + lh) * 224 + n0 + hi * 16];
      bf.q[0] = *(const f32x4*)(vp);
      bf.q[1] = *(const f32x4*)(vp + 8);
      acc = __builtin_amdgcn_wmma_f32_16x16x32_f16(false, a.v, false, bf.v,
                                                   (short)0, acc, false, false);
    }
    int e = et * 16 + lh;
#pragma unroll
    for (int v = 0; v < 8; ++v) {
      int j = jt * 16 + v + hi * 8;
      if ((j / MM) == (e >> 3)) {           // keep head-diagonal block
        int m = j % MM;
        float sv = acc[v];
        Slds[m * DD + e] = sv;
        out[OFF_S + (long)b * (MM * DD) + m * DD + e] = sv;
      }
    }
  }
  // A_maps store (Aavg finalized by previous barrier)
  for (int i = tid; i < MM * NN; i += 128)
    out[OFF_AM + (long)b * (MM * NN) + i] = Aavg[i];
  // top-9 mass per slot m via iterative wave-argmax extraction
  for (int m = wave; m < MM; m += 4) {
    float vals[7];
#pragma unroll
    for (int i = 0; i < 7; ++i) {
      int n = lane + 32 * i;
      vals[i] = (n < NN) ? Aavg[m * NN + n] : -1.0f;
    }
    float mass = 0.f;
    for (int it = 0; it < 9; ++it) {
      float lm = -2.f; int li = 0;
#pragma unroll
      for (int i = 0; i < 7; ++i)
        if (vals[i] > lm) { lm = vals[i]; li = i; }
      float bm = lm; int bl = lane;
#pragma unroll
      for (int o = 16; o; o >>= 1) {
        float ov = __shfl_xor(bm, o, 32);
        int   ol = __shfl_xor(bl, o, 32);
        if (ov > bm || (ov == bm && ol < bl)) { bm = ov; bl = ol; }
      }
      mass += bm;
      if (lane == bl) vals[li] = -2.f;
    }
    if (lane == 0) out[OFF_TK + (long)b * MM + m] = mass;
  }
  __syncthreads();

  // ---- Phase 4: Z = S_flat . Wp^T + bp ----------------------------------
  {
    int dout = tid & 31;
    int c0 = (tid >> 5) * 96;
    float z = 0.f;
#pragma unroll 8
    for (int k = 0; k < 96; ++k) z += Slds[c0 + k] * Wp[dout * 384 + c0 + k];
    atomicAdd(&Zacc[dout], z);
  }
  __syncthreads();
  if (tid < DD) out[OFF_Z + (long)b * DD + tid] = Zacc[tid];
}

extern "C" void kernel_launch(void* const* d_in, const int* in_sizes, int n_in,
                              void* d_out, int out_size, void* d_ws, size_t ws_size,
                              hipStream_t stream) {
  const float* T        = (const float*)d_in[0];
  const float* Q        = (const float*)d_in[1];
  const float* Wq       = (const float*)d_in[2];
  const float* Wk       = (const float*)d_in[3];
  const float* Wv       = (const float*)d_in[4];
  const float* Wp       = (const float*)d_in[5];
  const float* bp       = (const float*)d_in[6];
  const float* rpb_bias = (const float*)d_in[7];
  const float* qc       = (const float*)d_in[8];
  const float* kc       = (const float*)d_in[9];
  float* out = (float*)d_out;

  _Float16* qk   = (_Float16*)d_ws;                      // 48*32  f16 (3072 B)
  _Float16* wv16 = qk + 48 * 32;                         // 32*32  f16 (2048 B)
  float*    rpbp = (float*)((char*)d_ws + 5120);         // 48*208 f32 (39936 B)

  precompute_kernel<<<1, 256, 0, stream>>>(Q, Wq, Wk, Wv, qc, kc, rpb_bias,
                                           qk, wv16, rpbp);
  slotpool_main<<<BB, 128, 0, stream>>>(T, Wp, bp, qk, wv16, rpbp, out);
}
